// CubeNet2_86208583565580
// MI455X (gfx1250) — matrix operand order repaired and checked by reading, not compiled, not run on hardware
//
#include <hip/hip_runtime.h>

// ---------------------------------------------------------------------------
// CubeNet2 inference scan for gfx1250 (MI455X): wave32 + v_wmma_f32_16x16x32_f16
// H1pre kept in LDS (f16) — one persistent workgroup uses the 320KB WGP LDS.
// ---------------------------------------------------------------------------

typedef __attribute__((ext_vector_type(16))) _Float16 v16h;
typedef __attribute__((ext_vector_type(8)))  _Float16 v8h;
typedef __attribute__((ext_vector_type(8)))  float    v8f;

#define MGC    80
#define LSTM   500
#define KPAD   512     // K padded for WMMA (zeros beyond 500)
#define NP     2000    // gate columns (permuted p = 4*n + gate)
#define NTILE1 125     // NP/16 gate tiles
#define HID    300
#define OUTS   16
#define NB     4512    // 15*300 (MLP hidden) + 2 (out0) + 10 pad
#define NTILE3 282     // NB/16
#define BATCH  16
#define TF     128
#define TSTEPS 2048

__device__ __forceinline__ v8f wmma_f16(v16h a, v16h b, v8f c) {
  return __builtin_amdgcn_wmma_f32_16x16x32_f16(false, a, false, b, (short)0, c,
                                                false, false);
}

// A fragment: row-major f16 [16][stride]; caller passes p = base + m*stride + k0 + kb
// halves 0..7 <- p[0..7], halves 8..15 <- p[16..23]
__device__ __forceinline__ v16h loadA_frag(const _Float16* p) {
  v8h lo = *(const v8h*)(p);
  v8h hi = *(const v8h*)(p + 16);
  return __builtin_shufflevector(lo, hi, 0, 1, 2, 3, 4, 5, 6, 7,
                                 8, 9, 10, 11, 12, 13, 14, 15);
}

// B fragment: pre-transposed f16 [n][k]; caller passes p = row_n + k0 + ko
__device__ __forceinline__ v16h loadB_frag(const _Float16* p) {
  v8h lo = *(const v8h*)(p);
  v8h hi = *(const v8h*)(p + 16);
  return __builtin_shufflevector(lo, hi, 0, 1, 2, 3, 4, 5, 6, 7,
                                 8, 9, 10, 11, 12, 13, 14, 15);
}

__device__ __forceinline__ float sigm(float x) {
  return 1.0f / (1.0f + __expf(-x));
}

// permuted gate column p -> original column (gate*500 + n)
__device__ __forceinline__ int origcol(int p) { return (p & 3) * LSTM + (p >> 2); }

// ---------------------------------------------------------------------------
// Pack kernel: build f16 transposed weight tables (per launch).
// ---------------------------------------------------------------------------
__global__ void pack_kernel(const float* __restrict__ Wi,
                            const float* __restrict__ Wh,
                            const float* __restrict__ W1,
                            const float* __restrict__ W_out,
                            const float* __restrict__ b1,
                            const float* __restrict__ b_out,
                            _Float16* __restrict__ WhT,
                            _Float16* __restrict__ WiT,
                            _Float16* __restrict__ WxT,
                            _Float16* __restrict__ BigBT,
                            float* __restrict__ biasBig) {
  const int gid = blockIdx.x * blockDim.x + threadIdx.x;
  const int stride = gridDim.x * blockDim.x;

  for (int i = gid; i < NP * KPAD; i += stride) {
    int p = i >> 9, k = i & (KPAD - 1);
    int oc = origcol(p);
    float wh = 0.f, wi = 0.f;
    if (k < LSTM) {
      wh = Wh[k * NP + oc];
      wi = Wi[k * NP + oc];
    }
    WhT[i] = (_Float16)wh;
    WiT[i] = (_Float16)wi;
  }
  for (int i = gid; i < NP * 32; i += stride) {
    int p = i >> 5, k = i & 31;
    float v = (k < OUTS) ? Wi[(LSTM + k) * NP + origcol(p)] : 0.f;
    WxT[i] = (_Float16)v;
  }
  for (int i = gid; i < NB * KPAD; i += stride) {
    int q = i >> 9, k = i & (KPAD - 1);
    float v = 0.f;
    if (k < LSTM) {
      if (q < 15 * HID) {
        int ii = q / HID, jj = q % HID;
        v = W1[(ii * (LSTM + OUTS - 1) + k) * HID + jj];
      } else if (q < 15 * HID + 2) {
        v = W_out[k * 2 + (q - 15 * HID)];
      }
    }
    BigBT[i] = (_Float16)v;
  }
  for (int i = gid; i < NB; i += stride) {
    float v = 0.f;
    if (i < 15 * HID) v = b1[i];
    else if (i < 15 * HID + 2) v = b_out[i - 15 * HID];
    biasBig[i] = v;
  }
}

// ---------------------------------------------------------------------------
// cond[f][b][0:512] = mgc[b][f][:] @ up_W + up_b  (cols 500..511 zero)
// ---------------------------------------------------------------------------
__global__ void cond_kernel(const float* __restrict__ mgc,
                            const float* __restrict__ up_W,
                            const float* __restrict__ up_b,
                            float* __restrict__ cond) {
  const int f = blockIdx.x, b = blockIdx.y;
  __shared__ float sm[MGC];
  if (threadIdx.x < MGC) sm[threadIdx.x] = mgc[(b * TF + f) * MGC + threadIdx.x];
  __syncthreads();
  for (int n = threadIdx.x; n < KPAD; n += blockDim.x) {
    float acc = 0.f;
    if (n < LSTM) {
      acc = up_b[n];
      for (int k = 0; k < MGC; ++k) acc += sm[k] * up_W[k * LSTM + n];
    }
    cond[(size_t)(f * BATCH + b) * KPAD + n] = acc;
  }
}

// ---------------------------------------------------------------------------
// Gc[f][m][p] = b_lstm[origcol(p)] + cond[f][m][:] @ Wi_cond[:, origcol(p)]
// ---------------------------------------------------------------------------
__global__ __launch_bounds__(256) void gc_kernel(const float* __restrict__ cond,
                                                 const _Float16* __restrict__ WiT,
                                                 const float* __restrict__ b_lstm,
                                                 float* __restrict__ Gc) {
  const int f = blockIdx.x;
  const int lane = threadIdx.x & 31;
  const int wv = threadIdx.x >> 5;
  const int j = blockIdx.y * 8 + wv;
  if (j >= NTILE1) return;

  const int nl = lane & 15;
  const int m = lane & 15;
  const int kb = (lane < 16) ? 0 : 8;    // A half-K base
  const int ko = (lane < 16) ? 0 : 8;    // B half-K base
  const int hi8 = (lane < 16) ? 0 : 8;
  const int q = j * 16 + nl;

  const float bias = b_lstm[origcol(q)];
  v8f acc;
#pragma unroll
  for (int r = 0; r < 8; ++r) acc[r] = bias;

  const float* crow = cond + (size_t)f * BATCH * KPAD + m * KPAD;
  const _Float16* brow = WiT + (size_t)q * KPAD;

  for (int kk = 0; kk < KPAD / 32; ++kk) {
    const float* pa = crow + kk * 32 + kb;
    v16h a;
#pragma unroll
    for (int e = 0; e < 8; ++e) {
      a[e] = (_Float16)pa[e];
      a[e + 8] = (_Float16)pa[16 + e];
    }
    v16h b = loadB_frag(brow + kk * 32 + ko);
    acc = wmma_f16(a, b, acc);
  }
#pragma unroll
  for (int r = 0; r < 8; ++r)
    Gc[(size_t)f * BATCH * NP + (size_t)(r + hi8) * NP + q] = acc[r];
}

// ---------------------------------------------------------------------------
// Persistent sequential scan: one workgroup, 32 wave32 waves, t = 0..2047.
// Phase 1: gates = Gc[t] + x_prev@Wx + h@Wh (WMMA) -> LSTM pointwise
// Phase 2: H1s(LDS,f16) = [r] @ BigB + biasBig (15 MLP first layers + out0)
// Phase 3: sequential zz chain, H1s/zbuf in LDS, W1/W2 read from global
// ---------------------------------------------------------------------------
__global__ __launch_bounds__(1024, 1) void scan_kernel(
    const float* __restrict__ Gc, const float* __restrict__ cond,
    const _Float16* __restrict__ WhT, const _Float16* __restrict__ WxT,
    const _Float16* __restrict__ BigBT, const float* __restrict__ biasBig,
    const float* __restrict__ eps, const float* __restrict__ W1,
    const float* __restrict__ W2, const float* __restrict__ b2,
    float* __restrict__ out) {
  __shared__ __align__(16) _Float16 h16[2][BATCH * KPAD];  // double-buffered h
  __shared__ __align__(16) _Float16 r16[BATCH * KPAD];     // r = h + ct
  __shared__ __align__(16) _Float16 x16[BATCH * 32];       // x_prev (K padded 32)
  __shared__ __align__(16) _Float16 H1s[BATCH * NB];       // MLP pre-activations
  __shared__ float zbuf[BATCH][16];                        // chained z values
  __shared__ float o2buf[BATCH][2];                        // (mu, logvar)

  const int tid = threadIdx.x;
  const int lane = tid & 31;
  const int wv = tid >> 5;
  const int nl = lane & 15;
  const int kb = (lane < 16) ? 0 : 8;
  const int ko = (lane < 16) ? 0 : 8;
  const int hi8 = (lane < 16) ? 0 : 8;
  const int m_a = lane & 15;

  for (int i = tid; i < BATCH * KPAD; i += 1024) {
    h16[0][i] = (_Float16)0.f;
    h16[1][i] = (_Float16)0.f;
    r16[i] = (_Float16)0.f;
  }
  for (int i = tid; i < BATCH * 32; i += 1024) x16[i] = (_Float16)0.f;

  float creg[4][8];  // cell state, register resident (quad-redundant)
#pragma unroll
  for (int i = 0; i < 4; ++i)
#pragma unroll
    for (int r = 0; r < 8; ++r) creg[i][r] = 0.f;

  __syncthreads();

  for (int t = 0; t < TSTEPS; ++t) {
    const int f = t >> 4;
    const int rb = t & 1, wb = rb ^ 1;
    const float* gcf = Gc + (size_t)f * BATCH * NP;
    const float* cf = cond + (size_t)f * BATCH * KPAD;

    // ---- Phase 1: LSTM gates + pointwise ------------------------------
#pragma unroll
    for (int i = 0; i < 4; ++i) {
      const int j = wv + 32 * i;
      if (j < NTILE1) {
        const int q = j * 16 + nl;
        v8f acc;
#pragma unroll
        for (int r = 0; r < 8; ++r) acc[r] = gcf[(size_t)(r + hi8) * NP + q];

        const _Float16* brow = WhT + (size_t)q * KPAD;
        const _Float16* arow = &h16[rb][m_a * KPAD];
        for (int kk = 0; kk < KPAD / 32; ++kk) {
          v16h a = loadA_frag(arow + kk * 32 + kb);
          v16h b = loadB_frag(brow + kk * 32 + ko);
          acc = wmma_f16(a, b, acc);
        }
        {  // x_prev contribution (K = 32, zero padded)
          v16h a = loadA_frag(&x16[m_a * 32] + kb);
          v16h b = loadB_frag(WxT + (size_t)q * 32 + ko);
          acc = wmma_f16(a, b, acc);
        }

        // pointwise: quad lanes hold gates i,f,g,o of the same column n
        const int basel = lane & ~3;
        const int n = (j << 2) + (nl >> 2);
#pragma unroll
        for (int r = 0; r < 8; ++r) {
          float gi = __shfl(acc[r], basel + 0, 32);
          float gf = __shfl(acc[r], basel + 1, 32);
          float gg = __shfl(acc[r], basel + 2, 32);
          float go = __shfl(acc[r], basel + 3, 32);
          float cn = sigm(gf) * creg[i][r] + sigm(gi) * tanhf(gg);
          creg[i][r] = cn;
          float hn = sigm(go) * tanhf(cn);
          if ((lane & 3) == 0) {
            int m = r + hi8;
            float rv = hn + cf[m * KPAD + n];
            h16[wb][m * KPAD + n] = (_Float16)hn;
            r16[m * KPAD + n] = (_Float16)rv;
          }
        }
      }
    }
    __syncthreads();

    // ---- Phase 2: H1s = r @ BigB + bias (all 15 MLP L1 + out0) --------
    for (int i = 0; i < 9; ++i) {
      const int jt = wv + 32 * i;
      if (jt < NTILE3) {
        const int q = jt * 16 + nl;
        const float bias = biasBig[q];
        v8f acc;
#pragma unroll
        for (int r = 0; r < 8; ++r) acc[r] = bias;
        const _Float16* brow = BigBT + (size_t)q * KPAD;
        const _Float16* arow = &r16[m_a * KPAD];
        for (int kk = 0; kk < KPAD / 32; ++kk) {
          v16h a = loadA_frag(arow + kk * 32 + kb);
          v16h b = loadB_frag(brow + kk * 32 + ko);
          acc = wmma_f16(a, b, acc);
        }
#pragma unroll
        for (int r = 0; r < 8; ++r)
          H1s[(r + hi8) * NB + q] = (_Float16)acc[r];
      }
    }
    __syncthreads();

    // ---- Phase 3: sequential zz chain ---------------------------------
    const int b = wv >> 1;        // batch row for this wave
    const int comp = wv & 1;      // 0 = mu, 1 = logvar

    // stage 0: out0 = r@W_out + b_out already in H1s columns 4500/4501
    if (lane == 0) o2buf[b][comp] = (float)H1s[b * NB + 15 * HID + comp];
    __syncthreads();
    if (tid < BATCH) {
      float mu = o2buf[tid][0], lv = o2buf[tid][1];
      float z = mu + eps[(size_t)t * 256 + tid * 16 + 0] * __expf(fmaxf(lv, -7.f));
      zbuf[tid][0] = z;
      x16[tid * 32 + 0] = (_Float16)z;
      out[(size_t)tid * (TSTEPS * OUTS) + t * OUTS + 0] = z;
    }
    __syncthreads();

    for (int ii = 0; ii < OUTS - 1; ++ii) {
      float o2 = 0.f;
      const _Float16* h1row = &H1s[b * NB + ii * HID];
      for (int jj = lane; jj < HID; jj += 32) {
        float pre = (float)h1row[jj];
        for (int s = 0; s <= ii; ++s)
          pre += zbuf[b][s] * W1[(ii * (LSTM + OUTS - 1) + LSTM + s) * HID + jj];
        float hh = tanhf(pre);
        o2 += hh * W2[(ii * HID + jj) * 2 + comp];
      }
#pragma unroll
      for (int off = 16; off >= 1; off >>= 1) o2 += __shfl_down(o2, off, 32);
      if (lane == 0) o2buf[b][comp] = o2 + b2[ii * 2 + comp];
      __syncthreads();
      if (tid < BATCH) {
        float mu = o2buf[tid][0], lv = o2buf[tid][1];
        float z = mu +
                  eps[(size_t)t * 256 + tid * 16 + (ii + 1)] * __expf(fmaxf(lv, -7.f));
        if (ii + 1 < OUTS - 1) zbuf[tid][ii + 1] = z;
        x16[tid * 32 + (ii + 1)] = (_Float16)z;
        out[(size_t)tid * (TSTEPS * OUTS) + t * OUTS + (ii + 1)] = z;
      }
      __syncthreads();
    }
    // x16 now holds x_next; h16[wb] holds new h; creg holds new c.
  }
}

// ---------------------------------------------------------------------------
// Host launcher
// ---------------------------------------------------------------------------
extern "C" void kernel_launch(void* const* d_in, const int* in_sizes, int n_in,
                              void* d_out, int out_size, void* d_ws,
                              size_t ws_size, hipStream_t stream) {
  (void)in_sizes; (void)n_in; (void)out_size; (void)ws_size;

  const float* mgc   = (const float*)d_in[0];
  const float* eps   = (const float*)d_in[1];
  const float* up_W  = (const float*)d_in[2];
  const float* up_b  = (const float*)d_in[3];
  const float* Wi    = (const float*)d_in[4];
  const float* Wh    = (const float*)d_in[5];
  const float* b_lstm= (const float*)d_in[6];
  const float* W_out = (const float*)d_in[7];
  const float* b_out = (const float*)d_in[8];
  const float* W1    = (const float*)d_in[9];
  const float* b1    = (const float*)d_in[10];
  const float* W2    = (const float*)d_in[11];
  const float* b2    = (const float*)d_in[12];
  float* out = (float*)d_out;

  // workspace layout (256B aligned)
  char* ws = (char*)d_ws;
  size_t off = 0;
  auto take = [&](size_t bytes) {
    size_t r = off;
    off += (bytes + 255) & ~(size_t)255;
    return r;
  };
  float*    ws_cond  = (float*)   (ws + take((size_t)TF * BATCH * KPAD * 4));
  float*    ws_Gc    = (float*)   (ws + take((size_t)TF * BATCH * NP * 4));
  _Float16* ws_WhT   = (_Float16*)(ws + take((size_t)NP * KPAD * 2));
  _Float16* ws_WiT   = (_Float16*)(ws + take((size_t)NP * KPAD * 2));
  _Float16* ws_WxT   = (_Float16*)(ws + take((size_t)NP * 32 * 2));
  _Float16* ws_BigBT = (_Float16*)(ws + take((size_t)NB * KPAD * 2));
  float*    ws_bias  = (float*)   (ws + take((size_t)NB * 4));

  pack_kernel<<<512, 256, 0, stream>>>(Wi, Wh, W1, W_out, b1, b_out, ws_WhT,
                                       ws_WiT, ws_WxT, ws_BigBT, ws_bias);
  cond_kernel<<<dim3(TF, BATCH), 128, 0, stream>>>(mgc, up_W, up_b, ws_cond);
  gc_kernel<<<dim3(TF, 16), 256, 0, stream>>>(ws_cond, ws_WiT, b_lstm, ws_Gc);
  scan_kernel<<<1, 1024, 0, stream>>>(ws_Gc, ws_cond, ws_WhT, ws_WxT, ws_BigBT,
                                      ws_bias, eps, W1, W2, b2, out);
}
